// StackedFcFast_22428319220271
// MI455X (gfx1250) — matrix operand order repaired and checked by reading, not compiled, not run on hardware
//
#include <hip/hip_runtime.h>

// ---------------- problem constants (match reference) ----------------
#define NE      32          // experts
#define KSEL    4           // selected experts per token
#define INC     1024        // in channels
#define OUTC    1024        // out channels
#define BATCH_  4096
#define TOKENS  (BATCH_ * KSEL)   // 16384 row-tasks

// ---------------- tiling ----------------
#define BR      64          // rows per block tile
#define BC      64          // cols per block tile
#define BK      32          // k-depth per step (matches wmma 16x16x32)
#define NSTEP   (INC / BK)  // 32 k-steps
#define LDSP    40          // padded LDS row stride (ushorts): 80B, 16B-aligned, bank-conflict-free
#define MAX_TILES (TOKENS / BR + NE)   // worst-case sum of ceil(count_e/BR) = 288

// ---------------- workspace layout (ints) ----------------
#define WS_COUNTS 0     // [32]
#define WS_OFFS   64    // [33] exclusive token prefix
#define WS_TILE   128   // [33] exclusive row-tile prefix
#define WS_FILL   192   // [32] scatter cursors
#define WS_PERM   256   // [16384] tokens grouped by expert

typedef __attribute__((ext_vector_type(16))) __bf16      v16bf;
typedef __attribute__((ext_vector_type(8)))  float       v8f;
typedef __attribute__((ext_vector_type(4)))  unsigned int v4u;

union FragBF { v16bf bf; v4u u4[2]; };

// fp32 -> packed bf16 pair via CDNA5 V_FMA_MIXLO/MIXHI_BF16 (RNE cvt of fma(x,1,0))
__device__ __forceinline__ unsigned int pk2(float lo, float hi) {
  unsigned int r;
  asm("v_fma_mixlo_bf16 %0, %1, 1.0, 0" : "=v"(r) : "v"(lo));
  asm("v_fma_mixhi_bf16 %0, %1, 1.0, 0" : "+v"(r) : "v"(hi));
  return r;
}

// ---------------- phase 0: zero counters ----------------
__global__ void moe_zero(int* __restrict__ ws) {
  ws[threadIdx.x] = 0;   // 256 threads cover counts/offs/tile/fill
}

// ---------------- phase 1: histogram tokens per expert ----------------
__global__ void moe_count(const int* __restrict__ idx, int* __restrict__ ws) {
  int t = blockIdx.x * blockDim.x + threadIdx.x;
  if (t < TOKENS) atomicAdd(&ws[WS_COUNTS + idx[t]], 1);
}

// ---------------- phase 2: serial 32-wide exclusive scans ----------------
__global__ void moe_scan(int* __restrict__ ws) {
  if (threadIdx.x == 0) {
    int off = 0, toff = 0;
    for (int e = 0; e < NE; ++e) {
      ws[WS_OFFS + e] = off;
      ws[WS_TILE + e] = toff;
      int c = ws[WS_COUNTS + e];
      off  += c;
      toff += (c + BR - 1) / BR;
    }
    ws[WS_OFFS + NE] = off;
    ws[WS_TILE + NE] = toff;
  }
}

// ---------------- phase 3: scatter token ids grouped by expert ----------------
__global__ void moe_scatter(const int* __restrict__ idx, int* __restrict__ ws) {
  int t = blockIdx.x * blockDim.x + threadIdx.x;
  if (t < TOKENS) {
    int e = idx[t];
    int p = atomicAdd(&ws[WS_FILL + e], 1);
    ws[WS_PERM + ws[WS_OFFS + e] + p] = t;
  }
}

// ---------------- phase 4: grouped GEMM, bf16 WMMA, f32 accumulate ----------------
__global__ __launch_bounds__(256)
void moe_gemm(const float* __restrict__ x, const float* __restrict__ w,
              const float* __restrict__ bias, float* __restrict__ out,
              const int* __restrict__ ws) {
  __shared__ __align__(16) unsigned short lds_a[2][BR][LDSP];
  __shared__ __align__(16) unsigned short lds_b[2][BC][LDSP];
  __shared__ int tokS[BR];

  const int ty = blockIdx.y;
  if (ty >= ws[WS_TILE + NE]) return;     // uniform early-exit

  // find owning expert for this row-tile (32-entry linear scan)
  int e = 0;
  while (e < NE - 1 && ws[WS_TILE + e + 1] <= ty) ++e;
  const int localTile = ty - ws[WS_TILE + e];
  const int rowStart  = ws[WS_OFFS + e] + localTile * BR;
  int cnt = ws[WS_COUNTS + e] - localTile * BR;
  if (cnt > BR) cnt = BR;

  const int tid     = threadIdx.x;
  const int colBase = blockIdx.x * BC;
  const float* wE   = w + (size_t)e * INC * OUTC;

  if (tid < BR) tokS[tid] = (tid < cnt) ? ws[WS_PERM + rowStart + tid] : -1;
  __syncthreads();

  // cooperative-load mapping
  const int aRow = tid >> 2;            // 0..63  (x rows)
  const int aK   = (tid & 3) << 3;      // 0,8,16,24
  const int bK   = (tid >> 4) << 1;     // 0,2,...,30 (two consecutive k per thread)
  const int bN   = (tid & 15) << 2;     // 0,4,...,60 (four cols per thread)
  const int tokA = tokS[aRow];

  float4 ra0, ra1, rb0, rb1;
  auto loadRegs = [&](int kk) {
    if (tokA >= 0) {
      const float* xp = x + (size_t)(tokA >> 2) * INC + kk + aK;  // token -> batch row (KSEL=4)
      ra0 = *(const float4*)xp;
      ra1 = *(const float4*)(xp + 4);
    } else {
      ra0 = make_float4(0.f, 0.f, 0.f, 0.f);
      ra1 = ra0;
    }
    const float* wp = wE + (size_t)(kk + bK) * OUTC + colBase + bN;
    rb0 = *(const float4*)wp;           // k-row bK
    rb1 = *(const float4*)(wp + OUTC);  // k-row bK+1
    __builtin_prefetch(wp + BK * OUTC, 0, 1);   // global_prefetch_b8: next k-step weights
  };

  auto storeLds = [&](int buf) {
    // A tile: row-major, one b128 store of 8 bf16
    v4u pa;
    pa.x = pk2(ra0.x, ra0.y); pa.y = pk2(ra0.z, ra0.w);
    pa.z = pk2(ra1.x, ra1.y); pa.w = pk2(ra1.z, ra1.w);
    *(v4u*)&lds_a[buf][aRow][aK] = pa;
    // B tile: column-major; two consecutive k of one column pack into one DWORD
    unsigned short* pb = &lds_b[buf][bN][bK];
    *(unsigned int*)(pb)            = pk2(rb0.x, rb1.x);
    *(unsigned int*)(pb + LDSP)     = pk2(rb0.y, rb1.y);
    *(unsigned int*)(pb + 2 * LDSP) = pk2(rb0.z, rb1.z);
    *(unsigned int*)(pb + 3 * LDSP) = pk2(rb0.w, rb1.w);
  };

  // wave -> output sub-tile mapping: 8 waves = 4 row-groups x 2 col-pairs
  const int lane  = tid & 31;
  const int wv    = tid >> 5;
  const int wRow  = (wv >> 1) << 4;     // 0,16,32,48
  const int wCol  = (wv & 1) << 5;      // 0,32
  const int l16   = lane & 15;
  const int lhalf = lane >> 4;

  v8f c0 = {0.f, 0.f, 0.f, 0.f, 0.f, 0.f, 0.f, 0.f};
  v8f c1 = c0;

  loadRegs(0);
  int buf = 0;
  for (int s = 0; s < NSTEP; ++s) {
    storeLds(buf);
    __syncthreads();                     // double-buffered: single barrier per step
    if (s + 1 < NSTEP) loadRegs((s + 1) * BK);

    // A fragment (contiguous k = 16*lhalf + s, same order as B): two 16B chunks per lane
    FragBF af, bf0, bf1;
    const unsigned short* pa = &lds_a[buf][wRow + l16][lhalf << 4];
    af.u4[0] = *(const v4u*)pa;          // K = lhalf*16 + 0..7
    af.u4[1] = *(const v4u*)(pa + 8);    // K = lhalf*16 + 8..15
    // B fragments: 16 consecutive K of one column per lane (column-major LDS)
    const unsigned short* pb0 = &lds_b[buf][wCol + l16][lhalf << 4];
    bf0.u4[0] = *(const v4u*)pb0;
    bf0.u4[1] = *(const v4u*)(pb0 + 8);
    const unsigned short* pb1 = &lds_b[buf][wCol + 16 + l16][lhalf << 4];
    bf1.u4[0] = *(const v4u*)pb1;
    bf1.u4[1] = *(const v4u*)(pb1 + 8);

    c0 = __builtin_amdgcn_wmma_f32_16x16x32_bf16(false, af.bf, false, bf0.bf,
                                                 (short)0, c0, false, false);
    c1 = __builtin_amdgcn_wmma_f32_16x16x32_bf16(false, af.bf, false, bf1.bf,
                                                 (short)0, c1, false, false);
    buf ^= 1;
  }

  // epilogue: bias + relu + scatter rows to out[token]
  const float* bE   = bias + (size_t)e * OUTC;
  const int    colL = colBase + wCol + l16;
  #pragma unroll
  for (int i = 0; i < 8; ++i) {
    int r = wRow + (lhalf << 3) + i;     // C/D layout: M = i + 8*(lane/16)
    if (r < cnt) {
      int t = tokS[r];
      float v0 = c0[i] + bE[colL];
      float v1 = c1[i] + bE[colL + 16];
      float* op = out + (size_t)t * OUTC;
      op[colL]      = v0 > 0.f ? v0 : 0.f;
      op[colL + 16] = v1 > 0.f ? v1 : 0.f;
    }
  }
}

// ---------------- host entry ----------------
extern "C" void kernel_launch(void* const* d_in, const int* in_sizes, int n_in,
                              void* d_out, int out_size, void* d_ws, size_t ws_size,
                              hipStream_t stream) {
  (void)in_sizes; (void)n_in; (void)out_size; (void)ws_size;
  const float* x    = (const float*)d_in[0];
  const int*   idx  = (const int*)d_in[1];
  const float* w    = (const float*)d_in[2];
  const float* bias = (const float*)d_in[3];
  float* out = (float*)d_out;
  int*   ws  = (int*)d_ws;

  moe_zero<<<dim3(1), dim3(256), 0, stream>>>(ws);
  moe_count<<<dim3(TOKENS / 256), dim3(256), 0, stream>>>(idx, ws);
  moe_scan<<<dim3(1), dim3(32), 0, stream>>>(ws);
  moe_scatter<<<dim3(TOKENS / 256), dim3(256), 0, stream>>>(idx, ws);

  dim3 grid(OUTC / BC, MAX_TILES);
  moe_gemm<<<grid, dim3(256), 0, stream>>>(x, w, bias, out, ws);
}
